// GRU_62397284876921
// MI455X (gfx1250) — compile-verified, matrix operand order checked by reading, not run on hardware
//
#include <hip/hip_runtime.h>
#include <cmath>

// ---------------------------------------------------------------------------
// 2-layer GRU (B=64, S=512, D=H=1024) for MI455X / gfx1250.
//  - Parallel precompute of x @ Wih0^T with bf16 WMMA (v_wmma_f32_16x16x32_bf16)
//  - Per-step recurrent GEMMs in bf16 WMMA, weights L2-resident (24MB bf16)
//  - 64-wide K-tiles staged into double-buffered LDS via
//    GLOBAL_LOAD_ASYNC_TO_LDS_B128 (ASYNCcnt); straight-line async issue;
//    8 WMMAs per barrier; conflict-free padded LDS (72-elem row stride)
//  - f32 accumulation + f32 gate math; bf16 only for matrix operands
// ---------------------------------------------------------------------------

#define B_   64
#define S_   512
#define D_   1024
#define H_   1024
#define G3_  3072   // 3*H

#define BK_   64    // K elements staged per LDS buffer (2 WMMA K-steps)
#define STR_  72    // padded LDS row stride in elements (144B, conflict-free)

typedef __attribute__((ext_vector_type(16))) __bf16          v16bf;
typedef __attribute__((ext_vector_type(8)))  float           v8f;
typedef __attribute__((ext_vector_type(8)))  unsigned short  v8us;

__device__ __forceinline__ unsigned short f2bf(float f) {
  unsigned u = __builtin_bit_cast(unsigned, f);
  unsigned r = 0x7FFFu + ((u >> 16) & 1u);   // round-to-nearest-even
  return (unsigned short)((u + r) >> 16);
}

// 32B fragment load from LDS as two 16B-aligned halves -> 2x ds_load_b128
struct us16pair { v8us lo, hi; };
__device__ __forceinline__ v16bf ld_frag(const unsigned short* p) {
  us16pair t;
  t.lo = *(const v8us*)p;
  t.hi = *(const v8us*)(p + 8);
  return __builtin_bit_cast(v16bf, t);
}

// gfx1250 async copy: 16B global -> LDS, tracked by ASYNCcnt (no VGPR data)
__device__ __forceinline__ void async_copy16(unsigned lds_off, const void* src) {
  asm volatile("global_load_async_to_lds_b128 %0, %1, off"
               :: "v"(lds_off), "v"((unsigned long long)src)
               : "memory");
}
__device__ __forceinline__ void wait_async0() {
  asm volatile("s_wait_asynccnt 0x0" ::: "memory");
}

__global__ void cvt_f32_bf16(const float* __restrict__ src,
                             unsigned short* __restrict__ dst, size_t n) {
  size_t i = (size_t)blockIdx.x * blockDim.x + threadIdx.x;
  if (i < n) dst[i] = f2bf(src[i]);
}

// ---------------------------------------------------------------------------
// WMMA GEMM: C[M,N] = A[M,K](bf16) * W[N,K](bf16)^T + bias[N]
// Wave tile 32x32 (4 accumulators), workgroup 8 waves arranged WRxWC.
// ---------------------------------------------------------------------------
template<int WR, int WC>
__device__ __forceinline__ void gemm_body(
    const unsigned short* __restrict__ A, int lda,
    const unsigned short* __restrict__ W, int ldw,
    const float* __restrict__ bias,
    float* __restrict__ C, int ldc, int K,
    unsigned short* sA, unsigned short* sB,  // [2][AM][STR_], [2][BN][STR_]
    int bx, int by)
{
  constexpr int AM   = WR * 32;        // macro-tile rows
  constexpr int BN   = WC * 32;        // macro-tile cols
  constexpr int AOPS = AM * 8 / 256;   // 16B async ops per thread for A tile
  constexpr int BOPS = BN * 8 / 256;   // (row = 64 elems = 8 x 16B)
  const int tid  = threadIdx.x;
  const int lane = tid & 31;
  const int wave = tid >> 5;           // 0..7
  const int wrow = wave % WR;
  const int wcol = wave / WR;
  const int mBase = by * AM;
  const int nBase = bx * BN;
  const int half = lane >> 4;          // lane group 0/1 -> K halves
  const int l16  = lane & 15;
  const int koff = half * 16;

  // LDS byte offsets (flat LDS aperture: addr[31:0] == LDS offset)
  const unsigned sAoff = (unsigned)(size_t)(const void*)sA;
  const unsigned sBoff = (unsigned)(size_t)(const void*)sB;

  // Hoisted per-thread staging descriptors (straight-line async issue)
  const unsigned short* aSrc[AOPS]; unsigned aDst[AOPS];
  const unsigned short* bSrc[BOPS]; unsigned bDst[BOPS];
#pragma unroll
  for (int it = 0; it < AOPS; ++it) {
    int o = tid + it * 256, row = o >> 3, q = o & 7;
    aSrc[it] = A + (size_t)(mBase + row) * lda + q * 8;
    aDst[it] = sAoff + (unsigned)(row * STR_ + q * 8) * 2;
  }
#pragma unroll
  for (int it = 0; it < BOPS; ++it) {
    int o = tid + it * 256, row = o >> 3, q = o & 7;
    bSrc[it] = W + (size_t)(nBase + row) * ldw + q * 8;
    bDst[it] = sBoff + (unsigned)(row * STR_ + q * 8) * 2;
  }

  auto stage = [&](int buf, int k) {
    const unsigned ab = (unsigned)(buf * AM * STR_ * 2);
    const unsigned bb = (unsigned)(buf * BN * STR_ * 2);
#pragma unroll
    for (int it = 0; it < AOPS; ++it) {
      __builtin_prefetch(aSrc[it] + k + BK_, 0, 0);   // pull k+64 toward GL2
      async_copy16(aDst[it] + ab, aSrc[it] + k);
    }
#pragma unroll
    for (int it = 0; it < BOPS; ++it)
      async_copy16(bDst[it] + bb, bSrc[it] + k);
  };

  // Per-wave fragment element offsets inside a staged tile
  const int ar0 = (wrow * 32 +      l16) * STR_ + koff;
  const int ar1 = (wrow * 32 + 16 + l16) * STR_ + koff;
  const int br0 = (wcol * 32 +      l16) * STR_ + koff;
  const int br1 = (wcol * 32 + 16 + l16) * STR_ + koff;

  v8f acc00 = {}, acc01 = {}, acc10 = {}, acc11 = {};

  stage(0, 0);
  wait_async0();
  __syncthreads();

  int cur = 0;
  for (int k = 0; k < K; k += BK_) {
    const int nxt = cur ^ 1;
    if (k + BK_ < K) stage(nxt, k + BK_);   // overlap DMA with WMMA

    const int abase = cur * AM * STR_;
    const int bbase = cur * BN * STR_;
#pragma unroll
    for (int kk = 0; kk < BK_; kk += 32) {  // 2 K-steps -> 8 WMMAs per barrier
      v16bf a0 = ld_frag(&sA[abase + ar0 + kk]);
      v16bf a1 = ld_frag(&sA[abase + ar1 + kk]);
      v16bf b0 = ld_frag(&sB[bbase + br0 + kk]);
      v16bf b1 = ld_frag(&sB[bbase + br1 + kk]);
      acc00 = __builtin_amdgcn_wmma_f32_16x16x32_bf16(false, a0, false, b0, (short)0, acc00, false, false);
      acc01 = __builtin_amdgcn_wmma_f32_16x16x32_bf16(false, a0, false, b1, (short)0, acc01, false, false);
      acc10 = __builtin_amdgcn_wmma_f32_16x16x32_bf16(false, a1, false, b0, (short)0, acc10, false, false);
      acc11 = __builtin_amdgcn_wmma_f32_16x16x32_bf16(false, a1, false, b1, (short)0, acc11, false, false);
    }

    wait_async0();        // my async fills of buf[nxt] landed in LDS
    __syncthreads();      // everyone done filling nxt + done reading cur
    cur = nxt;
  }

  // C/D layout: VGPR v, lanes 0-15 -> (M=m0+v, N=lane); lanes 16-31 -> (M=m0+8+v)
  const int m0 = mBase + wrow * 32;
  const int n0 = nBase + wcol * 32;
  const int colA  = n0 + l16;
  const int colB  = n0 + 16 + l16;
  const float bvA = bias[colA];
  const float bvB = bias[colB];
  const int r0 = m0 + half * 8;
  const int r1 = m0 + 16 + half * 8;
#pragma unroll
  for (int v = 0; v < 8; ++v) {
    C[(size_t)(r0 + v) * ldc + colA] = acc00[v] + bvA;
    C[(size_t)(r0 + v) * ldc + colB] = acc01[v] + bvB;
    C[(size_t)(r1 + v) * ldc + colA] = acc10[v] + bvA;
    C[(size_t)(r1 + v) * ldc + colB] = acc11[v] + bvB;
  }
}

template<int WR, int WC>
__global__ __launch_bounds__(256) void gemm_wmma_k(
    const unsigned short* __restrict__ A, int lda,
    const unsigned short* __restrict__ W, int ldw,
    const float* __restrict__ bias,
    float* __restrict__ C, int ldc, int K)
{
  __shared__ unsigned short sA[2 * WR * 32 * STR_];
  __shared__ unsigned short sB[2 * WC * 32 * STR_];
  gemm_body<WR, WC>(A, lda, W, ldw, bias, C, ldc, K, sA, sB,
                    blockIdx.x, blockIdx.y);
}

// Two independent GEMMs (gi1 and gh1) in one launch, selected by blockIdx.z.
template<int WR, int WC>
__global__ __launch_bounds__(256) void gemm_wmma_dual(
    const unsigned short* __restrict__ A0, int lda0,
    const unsigned short* __restrict__ W0, int ldw0,
    const float* __restrict__ bias0, float* __restrict__ C0, int ldc0,
    const unsigned short* __restrict__ A1, int lda1,
    const unsigned short* __restrict__ W1, int ldw1,
    const float* __restrict__ bias1, float* __restrict__ C1, int ldc1,
    int K)
{
  __shared__ unsigned short sA[2 * WR * 32 * STR_];
  __shared__ unsigned short sB[2 * WC * 32 * STR_];
  if (blockIdx.z == 0)
    gemm_body<WR, WC>(A0, lda0, W0, ldw0, bias0, C0, ldc0, K, sA, sB,
                      blockIdx.x, blockIdx.y);
  else
    gemm_body<WR, WC>(A1, lda1, W1, ldw1, bias1, C1, ldc1, K, sA, sB,
                      blockIdx.x, blockIdx.y);
}

// ---------------------------------------------------------------------------
// Elementwise GRU gate kernels (f32 math)
// ---------------------------------------------------------------------------
__global__ void gru_init(const float* __restrict__ h0in,
                         float* __restrict__ hA, float* __restrict__ hB,
                         unsigned short* __restrict__ hAbf,
                         unsigned short* __restrict__ hBbf,
                         float* __restrict__ out_sum)
{
  int i = blockIdx.x * blockDim.x + threadIdx.x;   // 0..B*H-1
  float a = h0in[i];
  float b = h0in[(size_t)B_ * H_ + i];
  hA[i] = a; hB[i] = b;
  hAbf[i] = f2bf(a); hBbf[i] = f2bf(b);
  out_sum[i] = 0.0f;
}

__device__ __forceinline__ float sigm(float x) { return 1.0f / (1.0f + __expf(-x)); }

// Layer 0: gi comes from precomputed Gx0[(b*S+s)*3H + ...]
__global__ void gru_ew0(float* __restrict__ h, unsigned short* __restrict__ hbf,
                        const float* __restrict__ gx0, const float* __restrict__ gh,
                        const float* __restrict__ xmask, int s)
{
  int idx = blockIdx.x * blockDim.x + threadIdx.x;  // 0..B*H-1
  int b = idx >> 10, j = idx & (H_ - 1);
  const float* gxb = gx0 + ((size_t)b * S_ + s) * G3_;
  const float* ghb = gh + (size_t)b * G3_;
  float r = sigm(gxb[j] + ghb[j]);
  float z = sigm(gxb[H_ + j] + ghb[H_ + j]);
  float n = tanhf(gxb[2 * H_ + j] + r * ghb[2 * H_ + j]);
  float hv = h[idx];
  float m = xmask[b * S_ + s];
  float hnew = (m > 0.5f) ? ((1.0f - z) * n + z * hv) : hv;
  h[idx] = hnew;
  hbf[idx] = f2bf(hnew);
}

// Layer 1: gi from per-step buffer; accumulate masked top-layer output.
__global__ void gru_ew1(float* __restrict__ h, unsigned short* __restrict__ hbf,
                        const float* __restrict__ gi, const float* __restrict__ gh,
                        const float* __restrict__ xmask, int s,
                        float* __restrict__ out_sum)
{
  int idx = blockIdx.x * blockDim.x + threadIdx.x;
  int b = idx >> 10, j = idx & (H_ - 1);
  const float* gxb = gi + (size_t)b * G3_;
  const float* ghb = gh + (size_t)b * G3_;
  float r = sigm(gxb[j] + ghb[j]);
  float z = sigm(gxb[H_ + j] + ghb[H_ + j]);
  float n = tanhf(gxb[2 * H_ + j] + r * ghb[2 * H_ + j]);
  float hv = h[idx];
  float m = xmask[b * S_ + s];
  float hnew = (m > 0.5f) ? ((1.0f - z) * n + z * hv) : hv;
  h[idx] = hnew;
  hbf[idx] = f2bf(hnew);
  out_sum[idx] += hnew * m;   // reference: masked top-layer output per step
}

__global__ void gru_final(const float* __restrict__ out_sum,
                          const float* __restrict__ hA,
                          const float* __restrict__ hB,
                          float* __restrict__ out)
{
  int i = blockIdx.x * blockDim.x + threadIdx.x;
  // mean over concat(outputs[S], hidden[L]) -> divide by S+L = 514
  out[i] = (out_sum[i] + hA[i] + hB[i]) * (1.0f / 514.0f);
}

// ---------------------------------------------------------------------------
extern "C" void kernel_launch(void* const* d_in, const int* in_sizes, int n_in,
                              void* d_out, int out_size, void* d_ws, size_t ws_size,
                              hipStream_t stream)
{
  (void)in_sizes; (void)n_in; (void)out_size; (void)ws_size;

  const float* x    = (const float*)d_in[0];
  const float* xm   = (const float*)d_in[1];
  const float* h0   = (const float*)d_in[2];
  const float* Wih0 = (const float*)d_in[3];
  const float* Whh0 = (const float*)d_in[4];
  const float* bih0 = (const float*)d_in[5];
  const float* bhh0 = (const float*)d_in[6];
  const float* Wih1 = (const float*)d_in[7];
  const float* Whh1 = (const float*)d_in[8];
  const float* bih1 = (const float*)d_in[9];
  const float* bhh1 = (const float*)d_in[10];
  float* out = (float*)d_out;

  // Workspace carve-out
  char* ws = (char*)d_ws;
  size_t off = 0;
  auto alloc = [&](size_t bytes) -> char* {
    char* p = ws + off;
    off += (bytes + 255) & ~(size_t)255;
    return p;
  };
  unsigned short* xbf = (unsigned short*)alloc((size_t)B_ * S_ * D_ * 2);
  unsigned short* w0i = (unsigned short*)alloc((size_t)G3_ * D_ * 2);
  unsigned short* w0h = (unsigned short*)alloc((size_t)G3_ * H_ * 2);
  unsigned short* w1i = (unsigned short*)alloc((size_t)G3_ * H_ * 2);
  unsigned short* w1h = (unsigned short*)alloc((size_t)G3_ * H_ * 2);
  float* gx0 = (float*)alloc((size_t)B_ * S_ * G3_ * 4);   // precomputed gi, layer 0
  float* gh0 = (float*)alloc((size_t)B_ * G3_ * 4);
  float* gi1 = (float*)alloc((size_t)B_ * G3_ * 4);
  float* gh1 = (float*)alloc((size_t)B_ * G3_ * 4);
  float* hA  = (float*)alloc((size_t)B_ * H_ * 4);
  float* hB  = (float*)alloc((size_t)B_ * H_ * 4);
  unsigned short* hAbf = (unsigned short*)alloc((size_t)B_ * H_ * 2);
  unsigned short* hBbf = (unsigned short*)alloc((size_t)B_ * H_ * 2);
  float* osum = (float*)alloc((size_t)B_ * H_ * 4);

  // Phase 0: bf16 packing (weights stay [N,K] row-major -> contiguous K frags)
  {
    size_t nx = (size_t)B_ * S_ * D_;
    cvt_f32_bf16<<<dim3((unsigned)((nx + 255) / 256)), 256, 0, stream>>>(x, xbf, nx);
    size_t nw = (size_t)G3_ * D_;
    dim3 g((unsigned)((nw + 255) / 256));
    cvt_f32_bf16<<<g, 256, 0, stream>>>(Wih0, w0i, nw);
    cvt_f32_bf16<<<g, 256, 0, stream>>>(Whh0, w0h, nw);
    cvt_f32_bf16<<<g, 256, 0, stream>>>(Wih1, w1i, nw);
    cvt_f32_bf16<<<g, 256, 0, stream>>>(Whh1, w1h, nw);
  }
  gru_init<<<(B_ * H_) / 256, 256, 0, stream>>>(h0, hA, hB, hAbf, hBbf, osum);

  // Phase 1: Gx0 = x @ Wih0^T + bih0 ; M=B*S=32768, N=3072, K=1024
  // macro tile 128x64 (waves 4x2), grid 48 x 256
  gemm_wmma_k<4, 2><<<dim3(G3_ / 64, (B_ * S_) / 128), 256, 0, stream>>>(
      xbf, D_, w0i, D_, bih0, gx0, G3_, D_);

  // Phase 2: sequential recurrence. M=64 -> waves 2x4, macro 64x128, grid 24x1
  dim3 gRec(G3_ / 128, 1);
  dim3 gRec2(G3_ / 128, 1, 2);
  for (int s = 0; s < S_; ++s) {
    // layer 0: gh0 = hA @ Whh0^T + bhh0
    gemm_wmma_k<2, 4><<<gRec, 256, 0, stream>>>(hAbf, H_, w0h, H_, bhh0, gh0, G3_, H_);
    gru_ew0<<<(B_ * H_) / 256, 256, 0, stream>>>(hA, hAbf, gx0, gh0, xm, s);
    // layer 1: gi1 = hA_new @ Wih1^T + bih1 ; gh1 = hB @ Whh1^T + bhh1
    gemm_wmma_dual<2, 4><<<gRec2, 256, 0, stream>>>(
        hAbf, H_, w1i, H_, bih1, gi1, G3_,
        hBbf, H_, w1h, H_, bhh1, gh1, G3_, H_);
    gru_ew1<<<(B_ * H_) / 256, 256, 0, stream>>>(hB, hBbf, gi1, gh1, xm, s, osum);
  }

  // Phase 3: mean over S+L timeslices
  gru_final<<<(B_ * H_) / 256, 256, 0, stream>>>(osum, hA, hB, out);
}
